// DependencyBertMix_46316927320622
// MI455X (gfx1250) — matrix operand order repaired and checked by reading, not compiled, not run on hardware
//
#include <hip/hip_runtime.h>
#include <hip/hip_bf16.h>

// ---------------- types / helpers ----------------
typedef __attribute__((ext_vector_type(16))) __bf16 bf16x16;
typedef __attribute__((ext_vector_type(8)))  __bf16 bf16x8;
typedef __attribute__((ext_vector_type(8)))  float  f32x8;

union AFrag { bf16x16 v; bf16x8 h[2]; };

// A fragment (16x32 bf16, M x K): lane L = (m = L&15, hh = L>>4).
// Lane holds 16 contiguous K at 16*hh of row m (matches B fragment).
__device__ __forceinline__ bf16x16 load_a_frag(const __bf16* rowk, int hh) {
  AFrag a;
  a.h[0] = *(const bf16x8*)(rowk + 16 * hh);
  a.h[1] = *(const bf16x8*)(rowk + 16 * hh + 8);
  return a.v;
}

// B fragment (32x16 bf16, K x N) from column-major storage (WT[n][k]):
// lane L = (n = L&15, hh = L>>4) holds 16 contiguous K at 16*hh.
__device__ __forceinline__ bf16x16 load_b_frag(const __bf16* colk, int hh) {
  return *(const bf16x16*)(colk + 16 * hh);
}

__device__ __forceinline__ f32x8 wmma_bf16(bf16x16 a, bf16x16 b, f32x8 c) {
  return __builtin_amdgcn_wmma_f32_16x16x32_bf16(false, a, false, b,
                                                 (short)0, c, false, false);
}

#define NB 8
#define NH 12
#define NT 512
#define ND 64
#define NC 768

// ---------------- prep: f32 -> bf16 ----------------
__global__ void k_cvt_bf16(const float* __restrict__ src, __bf16* __restrict__ dst, int n) {
  int i = blockIdx.x * 256 + threadIdx.x;
  if (i < n) dst[i] = (__bf16)src[i];
}

// src [K][N] f32  ->  dst [N][K] bf16 (column-major for WMMA B operand)
__global__ void k_transpose_bf16(const float* __restrict__ src, __bf16* __restrict__ dst,
                                 int K, int N) {
  int i = blockIdx.x * 256 + threadIdx.x;
  if (i >= K * N) return;
  int n = i / K, k = i - n * K;
  dst[i] = (__bf16)src[(size_t)k * N + n];
}

// ---------------- QKV projection GEMM ----------------
// Xb [4096][768] bf16, WT [768][768] bf16 (col-major), bias f32[768].
// mode 0: out[b][h][t][d] bf16 ; mode 1: out[b][h][d][t] bf16 (v transposed).
// grid: (128 row-tiles of 32, 3 col-chunks of 256), block 128 (4 waves x 4 N-tiles).
// 2 M-tiles per wave: each B fragment feeds 2 WMMAs -> half the weight traffic.
__global__ __launch_bounds__(128) void k_qkv_gemm(const __bf16* __restrict__ Xb,
                                                  const __bf16* __restrict__ WT,
                                                  const float* __restrict__ bias,
                                                  __bf16* __restrict__ out, int mode) {
  const int wave = threadIdx.x >> 5, lane = threadIdx.x & 31;
  const int li = lane & 15, hh = lane >> 4;
  const int r0 = blockIdx.x * 32;
  const int n0 = blockIdx.y * 256 + wave * 64;

  f32x8 zero = {};
  f32x8 acc[2][4] = {{zero, zero, zero, zero}, {zero, zero, zero, zero}};
  for (int k0 = 0; k0 < NC; k0 += 32) {
    bf16x16 a0 = load_a_frag(Xb + (size_t)(r0 + li) * NC + k0, hh);
    bf16x16 a1 = load_a_frag(Xb + (size_t)(r0 + 16 + li) * NC + k0, hh);
#pragma unroll
    for (int j = 0; j < 4; ++j) {
      bf16x16 b = load_b_frag(WT + (size_t)(n0 + j * 16 + li) * NC + k0, hh);
      acc[0][j] = wmma_bf16(a0, b, acc[0][j]);
      acc[1][j] = wmma_bf16(a1, b, acc[1][j]);
    }
  }
#pragma unroll
  for (int mt = 0; mt < 2; ++mt) {
#pragma unroll
    for (int j = 0; j < 4; ++j) {
      const int n = n0 + j * 16 + li;          // output column (D layout: lane&15)
      const int head = n >> 6, d = n & 63;
      const float bv = bias[n];
#pragma unroll
      for (int r = 0; r < 8; ++r) {
        const int gr = r0 + mt * 16 + r + 8 * hh;  // output row (D layout: r + 8h)
        const int bb = gr >> 9, t = gr & 511;
        const float val = acc[mt][j][r] + bv;
        size_t off = (mode == 0)
          ? (((size_t)(bb * NH + head) * NT + t) * ND + d)
          : (((size_t)(bb * NH + head) * ND + d) * NT + t);
        out[off] = (__bf16)val;
      }
    }
  }
}

// ---------------- scores = q k^T / 8 ----------------
// grid: (32 t-tiles, 96 bh), block 256 (8 waves x 4 N-tiles = 512 keys).
__global__ __launch_bounds__(256) void k_scores(const __bf16* __restrict__ qb,
                                                const __bf16* __restrict__ kb,
                                                float* __restrict__ scores) {
  const int wave = threadIdx.x >> 5, lane = threadIdx.x & 31;
  const int li = lane & 15, hh = lane >> 4;
  const int bh = blockIdx.y;
  const int t0 = blockIdx.x * 16;
  const int n0 = wave * 64;
  const __bf16* qbase = qb + (size_t)bh * NT * ND;
  const __bf16* kbase = kb + (size_t)bh * NT * ND;

  f32x8 zero = {};
  f32x8 acc[4] = {zero, zero, zero, zero};
  for (int k0 = 0; k0 < ND; k0 += 32) {
    bf16x16 a = load_a_frag(qbase + (size_t)(t0 + li) * ND + k0, hh);
#pragma unroll
    for (int j = 0; j < 4; ++j) {
      bf16x16 b = load_b_frag(kbase + (size_t)(n0 + j * 16 + li) * ND + k0, hh);
      acc[j] = wmma_bf16(a, b, acc[j]);
    }
  }
#pragma unroll
  for (int j = 0; j < 4; ++j) {
    const int s = n0 + j * 16 + li;
#pragma unroll
    for (int r = 0; r < 8; ++r) {
      const int t = t0 + r + 8 * hh;
      scores[((size_t)bh * NT + t) * NT + s] = acc[j][r] * 0.125f;
    }
  }
}

// ---------------- fused gating MLP + softmax + probs@v ----------------
// grid: (16 t-tiles of 32 rows, 96 bh), block 256 (8 waves).
// 32-row M-tile halves W1T/W2T L2 traffic; each B fragment feeds 2 WMMAs.
__global__ __launch_bounds__(256) void k_gate(const float* __restrict__ scores,
                                              const float* __restrict__ mask,
                                              const float* __restrict__ depm,
                                              const float* __restrict__ ln_g,
                                              const float* __restrict__ ln_b,
                                              const __bf16* __restrict__ W1T,
                                              const float* __restrict__ b1,
                                              const __bf16* __restrict__ W2T,
                                              const float* __restrict__ b2,
                                              const __bf16* __restrict__ vT,
                                              float* __restrict__ out) {
  __shared__ __align__(16) unsigned char smemA[32 * 1024 * 2]; // lnS bf16[32][1024] / mixS f32[32][512]
  __shared__ __align__(16) unsigned char smemB[32 * 512 * 2];  // h1S bf16[32][512]  / pS  bf16[32][512]
  __shared__ float red1[512], red2[512];
  __shared__ float s_mu[32], s_rstd[32], s_stat[32];

  __bf16* lnS  = (__bf16*)smemA;
  float*  mixS = (float*)smemA;
  __bf16* h1S  = (__bf16*)smemB;
  __bf16* pS   = (__bf16*)smemB;

  const int tid = threadIdx.x;
  const int bh = blockIdx.y;
  const int b = bh / NH, h = bh - b * NH;
  const int t0 = blockIdx.x * 32;
  const float* sc  = scores + ((size_t)bh * NT + t0) * NT;   // [32][512]
  const float* dmb = depm + ((size_t)b * NT + t0) * NT;      // [32][512]
  const float* mk  = mask + (size_t)b * NT;

  const int row = tid >> 4;   // 0..15 (handles rows row and row+16)
  const int c   = tid & 15;

  // --- phase A: LayerNorm statistics over the 1024-wide concat rows ---
#pragma unroll
  for (int rr = 0; rr < 2; ++rr) {
    const int m = row + 16 * rr;
    float s1 = 0.f, s2 = 0.f;
    for (int n = c; n < NT; n += 16) {
      float s  = sc[m * NT + n];
      float sa = s + (1.0f - mk[n]) * -10000.0f;
      float dp = s * dmb[m * NT + n];
      s1 += sa + dp;
      s2 += sa * sa + dp * dp;
    }
    red1[m * 16 + c] = s1;
    red2[m * 16 + c] = s2;
  }
  __syncthreads();
  if (c == 0) {
#pragma unroll
    for (int rr = 0; rr < 2; ++rr) {
      const int m = row + 16 * rr;
      float S1 = 0.f, S2 = 0.f;
#pragma unroll
      for (int i = 0; i < 16; ++i) { S1 += red1[m * 16 + i]; S2 += red2[m * 16 + i]; }
      float mu  = S1 * (1.0f / 1024.0f);
      float var = S2 * (1.0f / 1024.0f) - mu * mu;
      s_mu[m] = mu;
      s_rstd[m] = rsqrtf(var + 1e-5f);
    }
  }
  __syncthreads();

  // --- phase B: normalized bf16 rows into LDS ---
#pragma unroll
  for (int rr = 0; rr < 2; ++rr) {
    const int m = row + 16 * rr;
    const float mu = s_mu[m], rs = s_rstd[m];
    for (int n = c; n < NT; n += 16) {
      float s  = sc[m * NT + n];
      float sa = s + (1.0f - mk[n]) * -10000.0f;
      float dp = s * dmb[m * NT + n];
      lnS[m * 1024 + n]       = (__bf16)(((sa - mu) * rs) * ln_g[n] + ln_b[n]);
      lnS[m * 1024 + 512 + n] = (__bf16)(((dp - mu) * rs) * ln_g[512 + n] + ln_b[512 + n]);
    }
  }
  __syncthreads();

  const int wave = tid >> 5, lane = tid & 31;
  const int li = lane & 15, hh = lane >> 4;
  const int n0 = wave * 64;
  f32x8 zero = {};

  // --- GEMM1: h1 = tanh(ln @ W1 + b1), K=1024, 2 M-tiles ---
  {
    f32x8 acc[2][4] = {{zero, zero, zero, zero}, {zero, zero, zero, zero}};
    for (int k0 = 0; k0 < 1024; k0 += 32) {
      bf16x16 a0 = load_a_frag(lnS + li * 1024 + k0, hh);
      bf16x16 a1 = load_a_frag(lnS + (16 + li) * 1024 + k0, hh);
#pragma unroll
      for (int j = 0; j < 4; ++j) {
        bf16x16 bb = load_b_frag(W1T + (size_t)(n0 + j * 16 + li) * 1024 + k0, hh);
        acc[0][j] = wmma_bf16(a0, bb, acc[0][j]);
        acc[1][j] = wmma_bf16(a1, bb, acc[1][j]);
      }
    }
#pragma unroll
    for (int mt = 0; mt < 2; ++mt) {
#pragma unroll
      for (int j = 0; j < 4; ++j) {
        const int n = n0 + j * 16 + li;
        const float bv = b1[n];
#pragma unroll
        for (int r = 0; r < 8; ++r)
          h1S[(mt * 16 + r + 8 * hh) * NT + n] = (__bf16)tanhf(acc[mt][j][r] + bv);
      }
    }
  }
  __syncthreads();

  // --- GEMM2: g = sigmoid(h1 @ W2 + b2), K=512, 2 M-tiles; mix into mixS ---
  {
    f32x8 acc[2][4] = {{zero, zero, zero, zero}, {zero, zero, zero, zero}};
    for (int k0 = 0; k0 < NT; k0 += 32) {
      bf16x16 a0 = load_a_frag(h1S + li * NT + k0, hh);
      bf16x16 a1 = load_a_frag(h1S + (16 + li) * NT + k0, hh);
#pragma unroll
      for (int j = 0; j < 4; ++j) {
        bf16x16 bb = load_b_frag(W2T + (size_t)(n0 + j * 16 + li) * NT + k0, hh);
        acc[0][j] = wmma_bf16(a0, bb, acc[0][j]);
        acc[1][j] = wmma_bf16(a1, bb, acc[1][j]);
      }
    }
#pragma unroll
    for (int mt = 0; mt < 2; ++mt) {
#pragma unroll
      for (int j = 0; j < 4; ++j) {
        const int n = n0 + j * 16 + li;
        const float bv = b2[n];
        const float biasv = (1.0f - mk[n]) * -10000.0f;
#pragma unroll
        for (int r = 0; r < 8; ++r) {
          const int m = mt * 16 + r + 8 * hh;
          float g  = 1.0f / (1.0f + __expf(-(acc[mt][j][r] + bv)));
          float s  = sc[m * NT + n];
          float sa = s + biasv;
          float dp = s * dmb[m * NT + n];
          mixS[m * NT + n] = g * sa + (1.0f - g) * dp;
        }
      }
    }
  }
  __syncthreads();

  // --- softmax over mixS rows -> pS (bf16) ---
#pragma unroll
  for (int rr = 0; rr < 2; ++rr) {
    const int m = row + 16 * rr;
    float lm = -3.4e38f;
    for (int n = c; n < NT; n += 16) lm = fmaxf(lm, mixS[m * NT + n]);
    red1[m * 16 + c] = lm;
  }
  __syncthreads();
  if (c == 0) {
#pragma unroll
    for (int rr = 0; rr < 2; ++rr) {
      const int m = row + 16 * rr;
      float M = red1[m * 16];
#pragma unroll
      for (int i = 1; i < 16; ++i) M = fmaxf(M, red1[m * 16 + i]);
      s_stat[m] = M;
    }
  }
  __syncthreads();
#pragma unroll
  for (int rr = 0; rr < 2; ++rr) {
    const int m = row + 16 * rr;
    const float M = s_stat[m];
    float ls = 0.f;
    for (int n = c; n < NT; n += 16) {
      float e = __expf(mixS[m * NT + n] - M);
      mixS[m * NT + n] = e;
      ls += e;
    }
    red1[m * 16 + c] = ls;
  }
  __syncthreads();
  if (c == 0) {
#pragma unroll
    for (int rr = 0; rr < 2; ++rr) {
      const int m = row + 16 * rr;
      float S = 0.f;
#pragma unroll
      for (int i = 0; i < 16; ++i) S += red1[m * 16 + i];
      s_stat[m] = 1.0f / S;
    }
  }
  __syncthreads();
#pragma unroll
  for (int rr = 0; rr < 2; ++rr) {
    const int m = row + 16 * rr;
    const float inv = s_stat[m];
    for (int n = c; n < NT; n += 16)
      pS[m * NT + n] = (__bf16)(mixS[m * NT + n] * inv);
  }
  __syncthreads();

  // --- GEMM3: ctx = probs @ v ---
  // 2 M-tiles x 4 D-tiles = 8 tile jobs; one per wave. K=512.
  {
    const int mt = wave >> 2;              // 0..1
    const int d0 = (wave & 3) * 16;        // 0..48
    f32x8 acc3 = zero;
    const __bf16* vb = vT + (size_t)bh * ND * NT;   // [d][s]
    for (int k0 = 0; k0 < NT; k0 += 32) {
      bf16x16 a = load_a_frag(pS + (mt * 16 + li) * NT + k0, hh);
      bf16x16 bb = load_b_frag(vb + (size_t)(d0 + li) * NT + k0, hh);
      acc3 = wmma_bf16(a, bb, acc3);
    }
    const int col = h * ND + d0 + li;
#pragma unroll
    for (int r = 0; r < 8; ++r) {
      const int t = t0 + mt * 16 + r + 8 * hh;
      out[((size_t)b * NT + t) * NC + col] = acc3[r];
    }
  }
}

// ---------------- launch ----------------
extern "C" void kernel_launch(void* const* d_in, const int* in_sizes, int n_in,
                              void* d_out, int out_size, void* d_ws, size_t ws_size,
                              hipStream_t stream) {
  (void)in_sizes; (void)n_in; (void)out_size; (void)ws_size;
  const float* hs   = (const float*)d_in[0];
  const float* mask = (const float*)d_in[1];
  const float* depm = (const float*)d_in[2];
  const float* Wq   = (const float*)d_in[3];
  const float* bq   = (const float*)d_in[4];
  const float* Wk   = (const float*)d_in[5];
  const float* bk   = (const float*)d_in[6];
  const float* Wv   = (const float*)d_in[7];
  const float* bv   = (const float*)d_in[8];
  const float* ln_g = (const float*)d_in[9];
  const float* ln_b = (const float*)d_in[10];
  const float* W1   = (const float*)d_in[11];
  const float* b1   = (const float*)d_in[12];
  const float* W2   = (const float*)d_in[13];
  const float* b2   = (const float*)d_in[14];
  float* out = (float*)d_out;

  char* ws = (char*)d_ws;
  size_t off = 0;
  auto carve = [&](size_t bytes) -> void* {
    void* p = ws + off;
    off = (off + bytes + 255) & ~(size_t)255;
    return p;
  };
  const size_t NROW = (size_t)NB * NT;          // 4096
  __bf16* Xb     = (__bf16*)carve(NROW * NC * 2);
  __bf16* WqT    = (__bf16*)carve((size_t)NC * NC * 2);
  __bf16* WkT    = (__bf16*)carve((size_t)NC * NC * 2);
  __bf16* WvT    = (__bf16*)carve((size_t)NC * NC * 2);
  __bf16* W1T    = (__bf16*)carve((size_t)512 * 1024 * 2);
  __bf16* W2T    = (__bf16*)carve((size_t)512 * 512 * 2);
  __bf16* qb     = (__bf16*)carve((size_t)NB * NH * NT * ND * 2);
  __bf16* kb     = (__bf16*)carve((size_t)NB * NH * NT * ND * 2);
  __bf16* vTb    = (__bf16*)carve((size_t)NB * NH * ND * NT * 2);
  float*  scores = (float*)carve((size_t)NB * NH * NT * NT * 4);

  // prep
  {
    int n = (int)(NROW * NC);
    k_cvt_bf16<<<(n + 255) / 256, 256, 0, stream>>>(hs, Xb, n);
  }
  {
    int n = NC * NC;
    k_transpose_bf16<<<(n + 255) / 256, 256, 0, stream>>>(Wq, WqT, NC, NC);
    k_transpose_bf16<<<(n + 255) / 256, 256, 0, stream>>>(Wk, WkT, NC, NC);
    k_transpose_bf16<<<(n + 255) / 256, 256, 0, stream>>>(Wv, WvT, NC, NC);
    int n1 = 1024 * 512;
    k_transpose_bf16<<<(n1 + 255) / 256, 256, 0, stream>>>(W1, W1T, 1024, 512);
    int n2 = 512 * 512;
    k_transpose_bf16<<<(n2 + 255) / 256, 256, 0, stream>>>(W2, W2T, 512, 512);
  }

  // QKV projections (32-row tiles)
  {
    dim3 grid(NROW / 32, NC / 256);
    k_qkv_gemm<<<grid, 128, 0, stream>>>(Xb, WqT, bq, qb, 0);
    k_qkv_gemm<<<grid, 128, 0, stream>>>(Xb, WkT, bk, kb, 0);
    k_qkv_gemm<<<grid, 128, 0, stream>>>(Xb, WvT, bv, vTb, 1);
  }

  // scores
  {
    dim3 grid(NT / 16, NB * NH);
    k_scores<<<grid, 256, 0, stream>>>(qb, kb, scores);
  }

  // fused gating + softmax + context (32-row tiles)
  {
    dim3 grid(NT / 32, NB * NH);
    k_gate<<<grid, 256, 0, stream>>>(scores, mask, depm, ln_g, ln_b,
                                     W1T, b1, W2T, b2, vTb, out);
  }
}
